// VQVAELayer_61186104099449
// MI455X (gfx1250) — matrix operand order repaired and checked by reading, not compiled, not run on hardware
//
#include <hip/hip_runtime.h>
#include <hip/hip_bf16.h>

typedef __attribute__((ext_vector_type(16))) __bf16 v16bf;
typedef __attribute__((ext_vector_type(8)))  float  v8f;

#define Nn 262144
#define Dd 64
#define Kk 1024
#define GAMMA 0.99f

// ---------------------------------------------------------------------------
// Prep: per codebook column k:
//   wsq[k]   = sum_d w[d][k]^2                       (f32, exact)
//   wT[k][d] = w[d][k]                               (f32 transposed codebook)
//   bstaged  = bf16 B-matrix in WMMA lane/element order:
//              tile t = k>>4, col n = k&15, step s = d>>5, dd = d&31
//              lane = ((dd>>3)&1)*16 + n, element e = (dd&7) + (dd>>4)*8
// ---------------------------------------------------------------------------
__global__ void vq_prep(const float* __restrict__ w,
                        __bf16* __restrict__ bstaged,
                        float* __restrict__ wT,
                        float* __restrict__ wsq) {
    int k = blockIdx.x * blockDim.x + threadIdx.x;
    if (k >= Kk) return;
    int t = k >> 4, n = k & 15;
    float s = 0.f;
    for (int d = 0; d < Dd; ++d) {
        float v = w[d * Kk + k];
        s += v * v;
        wT[k * Dd + d] = v;
        int st   = d >> 5;
        int dd   = d & 31;
        int lane = (((dd >> 3) & 1) << 4) + n;
        int e    = (dd & 7) + ((dd >> 4) << 3);
        bstaged[(((t * 2 + st) * 32 + lane) << 4) + e] = (__bf16)v;
    }
    wsq[k] = s;
}

// ---------------------------------------------------------------------------
// Main: each wave32 owns TWO 16-row tiles of x (32 rows); 8 waves / block
// -> 256 rows per block, grid = N/256 = 1024.
// bf16 WMMA (f32 acc) computes x.w; dist = ||w||^2 - 2 x.w (||x||^2 dropped,
// argmin-invariant). B tiles pre-swizzled in global, cycled through LDS in
// 32KB phases and shared by all 8 waves; wsq staged in LDS too.
// ---------------------------------------------------------------------------
__launch_bounds__(256)
__global__ void vq_assign(const float* __restrict__ x,
                          const __bf16* __restrict__ bstaged,
                          const float* __restrict__ wsq,
                          const float* __restrict__ wT,
                          float* __restrict__ quant,   // [N][64]
                          float* __restrict__ acc,     // [K][64]
                          float* __restrict__ cnt) {   // [K]
    __shared__ __bf16 sB[16384];   // 32 KB: 16 tiles x 2 steps x 32 lanes x 16
    __shared__ float  sWsq[Kk];    // 4 KB
    __shared__ int    sIdx[8][32];

    const int tid   = threadIdx.x;
    const int wave  = tid >> 5;
    const int lane  = tid & 31;
    const int m0    = (blockIdx.x * 8 + wave) * 32;
    const int rlo   = lane & 15;           // row within tile for A layout
    const int khalf = lane >> 4;           // which K half this lane holds
    const int kb    = khalf * 8;           // A-matrix K base for this half
    const int n_lane = rlo;                // C/D column this lane holds

    // stage wsq into LDS (visible after the phase-0 barrier)
    for (int i = tid; i < Kk; i += 256) sWsq[i] = wsq[i];

    // ---- load A for two row tiles (rows m0.. and m0+16..), D=64 ----
    const float* xr0 = x + (size_t)(m0 + rlo) * Dd;
    const float* xr1 = x + (size_t)(m0 + 16 + rlo) * Dd;
    v16bf a0, a1, a2, a3;
#pragma unroll
    for (int e = 0; e < 8; ++e) {
        a0[e]     = (__bf16)xr0[kb + e];
        a0[e + 8] = (__bf16)xr0[kb + 16 + e];
        a1[e]     = (__bf16)xr0[32 + kb + e];
        a1[e + 8] = (__bf16)xr0[32 + kb + 16 + e];
        a2[e]     = (__bf16)xr1[kb + e];
        a2[e + 8] = (__bf16)xr1[kb + 16 + e];
        a3[e]     = (__bf16)xr1[32 + kb + e];
        a3[e + 8] = (__bf16)xr1[32 + kb + 16 + e];
    }

    float minv0[8], minv1[8];
    int   mini0[8], mini1[8];
#pragma unroll
    for (int r = 0; r < 8; ++r) {
        minv0[r] = 3.4e38f; mini0[r] = 0;
        minv1[r] = 3.4e38f; mini1[r] = 0;
    }

    for (int phase = 0; phase < 4; ++phase) {
        // cooperative copy of 16 column-tiles (32 KB) into LDS
        const float4* src = (const float4*)(bstaged + phase * 16384);
        float4*       dst = (float4*)sB;
        for (int i = tid; i < 2048; i += 256) dst[i] = src[i];
        __syncthreads();

        for (int tl = 0; tl < 16; ++tl) {
            const v16bf* bp = (const v16bf*)(sB + ((tl * 2) * 32 + lane) * 16);
            v16bf b0 = bp[0];
            v16bf b1 = bp[32];   // step 1 is +512 bf16 elements

            // two independent accumulation chains (better XDL overlap)
            v8f c0 = {}, c1 = {};
            c0 = __builtin_amdgcn_wmma_f32_16x16x32_bf16(false, a0, false, b0,
                                                         (short)0, c0, false, false);
            c1 = __builtin_amdgcn_wmma_f32_16x16x32_bf16(false, a2, false, b0,
                                                         (short)0, c1, false, false);
            c0 = __builtin_amdgcn_wmma_f32_16x16x32_bf16(false, a1, false, b1,
                                                         (short)0, c0, false, false);
            c1 = __builtin_amdgcn_wmma_f32_16x16x32_bf16(false, a3, false, b1,
                                                         (short)0, c1, false, false);

            const int   n  = (phase * 16 + tl) * 16 + n_lane;
            const float wq = sWsq[n];
#pragma unroll
            for (int r = 0; r < 8; ++r) {
                float d0 = fmaf(-2.f, c0[r], wq);   // ||w||^2 - 2 x.w
                float d1 = fmaf(-2.f, c1[r], wq);
                if (d0 < minv0[r]) { minv0[r] = d0; mini0[r] = n; }
                if (d1 < minv1[r]) { minv1[r] = d1; mini1[r] = n; }
            }
        }
        __syncthreads();
    }

    // ---- reduce (min, idx) across the 16 lanes of each half ----
#pragma unroll
    for (int off = 1; off < 16; off <<= 1) {
#pragma unroll
        for (int r = 0; r < 8; ++r) {
            float ov = __shfl_xor(minv0[r], off, 32);
            int   oi = __shfl_xor(mini0[r], off, 32);
            if (ov < minv0[r] || (ov == minv0[r] && oi < mini0[r])) {
                minv0[r] = ov; mini0[r] = oi;
            }
            ov = __shfl_xor(minv1[r], off, 32);
            oi = __shfl_xor(mini1[r], off, 32);
            if (ov < minv1[r] || (ov == minv1[r] && oi < mini1[r])) {
                minv1[r] = ov; mini1[r] = oi;
            }
        }
    }
    if (rlo == 0) {   // lanes 0 and 16 hold the reduced results
#pragma unroll
        for (int r = 0; r < 8; ++r) {
            sIdx[wave][khalf * 8 + r]      = mini0[r];
            sIdx[wave][16 + khalf * 8 + r] = mini1[r];
        }
    }
    __syncthreads();

    // ---- gather quantized rows (exact f32) + scatter segment sums ----
    // one row per lane: row = m0 + lane
    const int    idx  = sIdx[wave][lane];
    const size_t mrow = (size_t)(m0 + lane);

    const float4* wsrc = (const float4*)(wT + (size_t)idx * Dd);
    float4*       qdst = (float4*)(quant + mrow * Dd);
#pragma unroll
    for (int j = 0; j < 16; ++j) qdst[j] = wsrc[j];

    const float* xsrc = x + mrow * Dd;
    float*       ab   = acc + (size_t)idx * Dd;
#pragma unroll
    for (int j = 0; j < 64; ++j) atomicAdd(ab + j, xsrc[j]);
    atomicAdd(cnt + idx, 1.0f);
}

// ---------------------------------------------------------------------------
// Finalize: EMA update + divide -> new_w [D][K]
// ---------------------------------------------------------------------------
__global__ void vq_final(const float* __restrict__ centroid_sum, // [D][K]
                         const float* __restrict__ centroid_n,   // [K]
                         const float* __restrict__ acc,          // [K][D]
                         const float* __restrict__ cnt,          // [K]
                         float* __restrict__ new_w) {            // [D][K]
    int i = blockIdx.x * 256 + threadIdx.x;   // 0..65535
    int d = i >> 10;
    int k = i & 1023;
    float cs = centroid_sum[i] * GAMMA + acc[k * Dd + d] * (1.0f - GAMMA);
    float cn = centroid_n[k]  * GAMMA + cnt[k]           * (1.0f - GAMMA);
    new_w[i] = cs / cn;
}

extern "C" void kernel_launch(void* const* d_in, const int* in_sizes, int n_in,
                              void* d_out, int out_size, void* d_ws, size_t ws_size,
                              hipStream_t stream) {
    const float* x    = (const float*)d_in[0];   // [N, 64]
    const float* w    = (const float*)d_in[1];   // [64, K]
    const float* csum = (const float*)d_in[2];   // [64, K]
    const float* cn   = (const float*)d_in[3];   // [K]

    float* quant = (float*)d_out;                            // [N, 64]
    float* new_w = (float*)d_out + (size_t)Nn * Dd;          // [64, K]

    char* ws = (char*)d_ws;
    __bf16* bstaged = (__bf16*)ws;                                   // 131072 B
    float*  wT      = (float*)(ws + 131072);                         // 262144 B
    float*  wsq     = (float*)(ws + 131072 + 262144);                //   4096 B
    float*  acc     = (float*)(ws + 131072 + 262144 + 4096);         // 262144 B
    float*  cnt     = (float*)(ws + 131072 + 262144 + 4096 + 262144);//   4096 B

    // zero segment-sum accumulators (acc + cnt are contiguous)
    hipMemsetAsync(acc, 0, 262144 + 4096, stream);

    vq_prep  <<<4,    256, 0, stream>>>(w, bstaged, wT, wsq);
    vq_assign<<<1024, 256, 0, stream>>>(x, bstaged, wsq, wT, quant, acc, cnt);
    vq_final <<<256,  256, 0, stream>>>(csum, cn, acc, cnt, new_w);
}
